// LSTM_Base_68135361184101
// MI455X (gfx1250) — matrix operand order repaired and checked by reading, not compiled, not run on hardware
//
#include <hip/hip_runtime.h>
#include <hip/hip_bf16.h>

#define T_STEPS 512
#define BATCH   64
#define INDIM   128
#define HDIM    512

typedef __bf16 bf16;
typedef __attribute__((ext_vector_type(16))) __bf16 v16bf;
typedef __attribute__((ext_vector_type(8)))  float  v8f;

// Dynamic-LDS layout (bytes). Weight regions hold WMMA B-fragments laid out
// [nt][kstep][lane][16 bf16] so each lane reads 32 contiguous bytes.
#define OFF_WIH0  0u         // 2 * 4  * 32 * 32 =  8192
#define OFF_WHH0  8192u      // 2 * 16 * 32 * 32 = 32768
#define OFF_WIH1  40960u     // 32768
#define OFF_WHH1  73728u     // 32768
#define OFF_GATES 106496u    // 2 kh-parts * 64 * 32 * 4 = 16384
#define OFF_C0    122880u    // 64 * 8 * 4 = 2048
#define OFF_C1    124928u    // 2048
#define OFF_B0    126976u    // 128
#define OFF_B1    127104u    // 128
#define SMEM_BYTES 127232u

union FragB16 { v16bf v; uint4 q[2]; };

// A operand (16x32 bf16, row-major source, row stride ld):
// lane L: row = L&15, half = L>>4. elems[0..15] = K(half*16 + 0..15)
// (contiguous, matching the B fragments in LDS)
__device__ inline v16bf load_a_frag(const bf16* __restrict__ base, int row, int ld,
                                    int k0, int lane) {
  const bf16* p = base + (size_t)row * ld + k0 + ((lane >> 4) << 4);
  FragB16 f;
  f.q[0] = *(const uint4*)(p);
  f.q[1] = *(const uint4*)(p + 8);
  return f.v;
}

// B operand fragment from LDS (fragment-ordered: 32B per lane)
__device__ inline v16bf load_b_lds(const char* p) {
  FragB16 f;
  f.q[0] = ((const uint4*)p)[0];
  f.q[1] = ((const uint4*)p)[1];
  return f.v;
}

__device__ inline v8f wmma_bf16(v16bf a, v16bf b, v8f c) {
  return __builtin_amdgcn_wmma_f32_16x16x32_bf16(false, a, false, b,
                                                 (short)0, c, false, false);
}

__device__ inline float fsigmoid(float x) { return 1.0f / (1.0f + __expf(-x)); }
__device__ inline float ftanh(float x) {
  x = fminf(fmaxf(x, -30.0f), 30.0f);
  float e = __expf(-2.0f * x);
  return (1.0f - e) / (1.0f + e);
}

__device__ inline void grid_barrier(unsigned* cnt, unsigned* gen, unsigned nwg) {
  __syncthreads();
  if (threadIdx.x == 0) {
    unsigned g = __hip_atomic_load(gen, __ATOMIC_RELAXED, __HIP_MEMORY_SCOPE_AGENT);
    unsigned prev = __hip_atomic_fetch_add(cnt, 1u, __ATOMIC_ACQ_REL,
                                           __HIP_MEMORY_SCOPE_AGENT);
    if (prev == nwg - 1u) {
      __hip_atomic_store(cnt, 0u, __ATOMIC_RELAXED, __HIP_MEMORY_SCOPE_AGENT);
      __hip_atomic_fetch_add(gen, 1u, __ATOMIC_RELEASE, __HIP_MEMORY_SCOPE_AGENT);
    } else {
      while (__hip_atomic_load(gen, __ATOMIC_ACQUIRE, __HIP_MEMORY_SCOPE_AGENT) == g) {
        __builtin_amdgcn_s_sleep(1);
      }
    }
  }
  __syncthreads();
}

__global__ void cvt_f32_bf16(const float* __restrict__ s, bf16* __restrict__ d, int n) {
  int i = blockIdx.x * blockDim.x + threadIdx.x;
  if (i < n) d[i] = (bf16)s[i];
}

// Copy this WG's B-fragments of one weight matrix into LDS, fragment-ordered.
// Entry idx = (nt*KS + ks)*32 + lane; each entry is 32B (16 bf16) of
// W[row(nt,lane)][ks*32 + (lane>>4)*16 ...].
__device__ inline void preload_w(const bf16* __restrict__ W, int Kdim,
                                 char* dstBase, int col0, int tid, int nthreads) {
  const int KS = Kdim >> 5;
  const int total = 2 * KS * 32;
  for (int idx = tid; idx < total; idx += nthreads) {
    int lane2 = idx & 31;
    int rest  = idx >> 5;                // nt*KS + ks
    int nt2   = rest / KS;
    int ks2   = rest % KS;
    int n2    = lane2 & 15;
    int row   = (nt2 * 2 + (n2 >> 3)) * HDIM + col0 + (n2 & 7);
    int koff  = (ks2 << 5) + ((lane2 >> 4) << 4);
    const uint4* src = (const uint4*)(W + (size_t)row * Kdim + koff);
    uint4* dst = (uint4*)(dstBase + (size_t)idx * 32);
    dst[0] = src[0];
    dst[1] = src[1];
  }
}

// Persistent 2-layer LSTM. Grid = 64 WGs x 256 threads (8 waves).
// WG owns hidden columns [8*blk, 8*blk+8) -> 32 gate columns = 2 N-tiles.
// Wave w = (mt = w>>1, kh = w&1): computes BOTH 16x16 N-tiles for batch rows
// 16*mt.. over K-half kh (disjoint A loads across waves; partial sums are
// reduced through two LDS gate buffers). Weights live in LDS all sequence.
__global__ __launch_bounds__(256) void lstm_persistent(
    const bf16* __restrict__ xbf,
    const bf16* __restrict__ Wih0, const bf16* __restrict__ Whh0,
    const bf16* __restrict__ Wih1, const bf16* __restrict__ Whh1,
    const float* __restrict__ bih0, const float* __restrict__ bhh0,
    const float* __restrict__ bih1, const float* __restrict__ bhh1,
    bf16* __restrict__ h0buf, bf16* __restrict__ h1buf,
    unsigned* __restrict__ bar_cnt, unsigned* __restrict__ bar_gen,
    float* __restrict__ out, int nwg) {
  extern __shared__ char smem[];
  float* s_g0    = (float*)(smem + OFF_GATES);          // [64][32] partial kh=0
  float* s_g1    = (float*)(smem + OFF_GATES + 8192u);  // [64][32] partial kh=1
  float* s_c0    = (float*)(smem + OFF_C0);             // [64][8]
  float* s_c1    = (float*)(smem + OFF_C1);
  float* s_bias0 = (float*)(smem + OFF_B0);             // [32] = 4 gates x 8 cols
  float* s_bias1 = (float*)(smem + OFF_B1);

  const int tid  = threadIdx.x;
  const int lane = tid & 31;
  const int wave = tid >> 5;
  const int mt   = wave >> 1;  // batch tile 0..3
  const int kh   = wave & 1;   // K-half 0..1
  const int n    = lane & 15;
  const int col0 = blockIdx.x * 8;

  // ---- one-time: biases, c-state, LDS-resident weights ----
  if (tid < 32) {
    int g = tid >> 3, c = tid & 7;
    int gr = g * HDIM + col0 + c;
    s_bias0[tid] = bih0[gr] + bhh0[gr];
    s_bias1[tid] = bih1[gr] + bhh1[gr];
  }
  for (int i = tid; i < BATCH * 8; i += blockDim.x) { s_c0[i] = 0.0f; s_c1[i] = 0.0f; }
  preload_w(Wih0, INDIM, smem + OFF_WIH0, col0, tid, 256);
  preload_w(Whh0, HDIM,  smem + OFF_WHH0, col0, tid, 256);
  preload_w(Wih1, HDIM,  smem + OFF_WIH1, col0, tid, 256);
  preload_w(Whh1, HDIM,  smem + OFF_WHH1, col0, tid, 256);
  __syncthreads();

  const int rowA = mt * 16 + n;   // batch row for A fragments
  // per-wave LDS fragment base pointers (index with (nt*KS + ks) << 10)
  const char* bWih0 = smem + OFF_WIH0 + ((unsigned)lane << 5);
  const char* bWhh0 = smem + OFF_WHH0 + ((unsigned)lane << 5);
  const char* bWih1 = smem + OFF_WIH1 + ((unsigned)lane << 5);
  const char* bWhh1 = smem + OFF_WHH1 + ((unsigned)lane << 5);
  float* gp = kh ? s_g1 : s_g0;   // this wave's partial-sum gate buffer

  for (int t = 0; t < T_STEPS; ++t) {
    const int cur = t & 1, prv = cur ^ 1;

    // ---------------- layer 0: gates = x_t W_ih0^T + h0 W_hh0^T ----------------
    v8f a0 = {}, a1 = {};
    const bf16* xt  = xbf + (size_t)t * BATCH * INDIM;
    const bf16* h0p = h0buf + (size_t)prv * BATCH * HDIM;
#pragma unroll
    for (int i = 0; i < 2; ++i) {          // x: K-half = 64 -> 2 ksteps
      int ks = kh * 2 + i;
      v16bf af = load_a_frag(xt, rowA, INDIM, ks * 32, lane);
      a0 = wmma_bf16(af, load_b_lds(bWih0 + ((unsigned)ks << 10)), a0);
      a1 = wmma_bf16(af, load_b_lds(bWih0 + ((unsigned)(4 + ks) << 10)), a1);
    }
#pragma unroll 4
    for (int i = 0; i < 8; ++i) {          // h: K-half = 256 -> 8 ksteps
      int ks = kh * 8 + i;
      v16bf af = load_a_frag(h0p, rowA, HDIM, ks * 32, lane);
      a0 = wmma_bf16(af, load_b_lds(bWhh0 + ((unsigned)ks << 10)), a0);
      a1 = wmma_bf16(af, load_b_lds(bWhh0 + ((unsigned)(16 + ks) << 10)), a1);
    }
    {
      // C/D layout: lane holds col n, rows (lane>>4)*8 + r in acc[r]
      int r0 = mt * 16 + ((lane >> 4) << 3);
#pragma unroll
      for (int r = 0; r < 8; ++r) {
        gp[(r0 + r) * 32 +      n] = a0[r];   // nt=0 cols
        gp[(r0 + r) * 32 + 16 + n] = a1[r];   // nt=1 cols
      }
    }
    __syncthreads();

    bf16* h0w = h0buf + (size_t)cur * BATCH * HDIM;
#pragma unroll
    for (int e = 0; e < 2; ++e) {
      int j = tid + e * 256;            // 512 = 64 rows x 8 cols
      int b = j >> 3, c = j & 7;
      float gi = s_g0[b * 32 +      c] + s_g1[b * 32 +      c] + s_bias0[     c];
      float gf = s_g0[b * 32 +  8 + c] + s_g1[b * 32 +  8 + c] + s_bias0[ 8 + c];
      float gg = s_g0[b * 32 + 16 + c] + s_g1[b * 32 + 16 + c] + s_bias0[16 + c];
      float go = s_g0[b * 32 + 24 + c] + s_g1[b * 32 + 24 + c] + s_bias0[24 + c];
      float cn = fsigmoid(gf) * s_c0[b * 8 + c] + fsigmoid(gi) * ftanh(gg);
      float hn = fsigmoid(go) * ftanh(cn);
      s_c0[b * 8 + c] = cn;
      h0w[b * HDIM + col0 + c] = (bf16)hn;
    }

    // one device-wide barrier per timestep (ping-pong h makes it sufficient)
    grid_barrier(bar_cnt, bar_gen, (unsigned)nwg);

    // ---------------- layer 1: gates = h0 W_ih1^T + h1 W_hh1^T ----------------
    v8f b0 = {}, b1 = {};
    const bf16* h0c = h0buf + (size_t)cur * BATCH * HDIM;
    const bf16* h1p = h1buf + (size_t)prv * BATCH * HDIM;
#pragma unroll 4
    for (int i = 0; i < 8; ++i) {
      int ks = kh * 8 + i;
      v16bf af = load_a_frag(h0c, rowA, HDIM, ks * 32, lane);
      b0 = wmma_bf16(af, load_b_lds(bWih1 + ((unsigned)ks << 10)), b0);
      b1 = wmma_bf16(af, load_b_lds(bWih1 + ((unsigned)(16 + ks) << 10)), b1);
    }
#pragma unroll 4
    for (int i = 0; i < 8; ++i) {
      int ks = kh * 8 + i;
      v16bf af = load_a_frag(h1p, rowA, HDIM, ks * 32, lane);
      b0 = wmma_bf16(af, load_b_lds(bWhh1 + ((unsigned)ks << 10)), b0);
      b1 = wmma_bf16(af, load_b_lds(bWhh1 + ((unsigned)(16 + ks) << 10)), b1);
    }
    {
      int r0 = mt * 16 + ((lane >> 4) << 3);
#pragma unroll
      for (int r = 0; r < 8; ++r) {
        gp[(r0 + r) * 32 +      n] = b0[r];
        gp[(r0 + r) * 32 + 16 + n] = b1[r];
      }
    }
    __syncthreads();

    bf16* h1w = h1buf + (size_t)cur * BATCH * HDIM;
#pragma unroll
    for (int e = 0; e < 2; ++e) {
      int j = tid + e * 256;
      int b = j >> 3, c = j & 7;
      float gi = s_g0[b * 32 +      c] + s_g1[b * 32 +      c] + s_bias1[     c];
      float gf = s_g0[b * 32 +  8 + c] + s_g1[b * 32 +  8 + c] + s_bias1[ 8 + c];
      float gg = s_g0[b * 32 + 16 + c] + s_g1[b * 32 + 16 + c] + s_bias1[16 + c];
      float go = s_g0[b * 32 + 24 + c] + s_g1[b * 32 + 24 + c] + s_bias1[24 + c];
      float cn = fsigmoid(gf) * s_c1[b * 8 + c] + fsigmoid(gi) * ftanh(gg);
      float hn = fsigmoid(go) * ftanh(cn);
      s_c1[b * 8 + c] = cn;
      h1w[b * HDIM + col0 + c] = (bf16)hn;
      out[((size_t)t * BATCH + b) * HDIM + col0 + c] = hn;
    }
    __syncthreads();  // protect gate buffers before next iteration's stores
  }
}

extern "C" void kernel_launch(void* const* d_in, const int* in_sizes, int n_in,
                              void* d_out, int out_size, void* d_ws, size_t ws_size,
                              hipStream_t stream) {
  const float* x     = (const float*)d_in[0];
  const float* Wih0f = (const float*)d_in[1];
  const float* Whh0f = (const float*)d_in[2];
  const float* bih0  = (const float*)d_in[3];
  const float* bhh0  = (const float*)d_in[4];
  const float* Wih1f = (const float*)d_in[5];
  const float* Whh1f = (const float*)d_in[6];
  const float* bih1  = (const float*)d_in[7];
  const float* bhh1  = (const float*)d_in[8];

  char* ws = (char*)d_ws;
  size_t off = 0;
  auto carve = [&](size_t bytes) -> void* {
    void* p = ws + off;
    off = (off + bytes + 255) & ~(size_t)255;
    return p;
  };

  unsigned* bar = (unsigned*)carve(256);                       // cnt, gen
  bf16* h0buf   = (bf16*)carve((size_t)2 * BATCH * HDIM * 2);  // ping-pong
  bf16* h1buf   = (bf16*)carve((size_t)2 * BATCH * HDIM * 2);
  bf16* Wih0b   = (bf16*)carve((size_t)4 * HDIM * INDIM * 2);
  bf16* Whh0b   = (bf16*)carve((size_t)4 * HDIM * HDIM * 2);
  bf16* Wih1b   = (bf16*)carve((size_t)4 * HDIM * HDIM * 2);
  bf16* Whh1b   = (bf16*)carve((size_t)4 * HDIM * HDIM * 2);
  bf16* xbf     = (bf16*)carve((size_t)T_STEPS * BATCH * INDIM * 2);

  // zero barrier state + h ping-pong buffers (deterministic per call)
  hipMemsetAsync(ws, 0, 256 + (size_t)2 * 2 * BATCH * HDIM * 2, stream);

  auto cvt = [&](const float* s, bf16* d, int nelem) {
    cvt_f32_bf16<<<(nelem + 255) / 256, 256, 0, stream>>>(s, d, nelem);
  };
  cvt(Wih0f, Wih0b, 4 * HDIM * INDIM);
  cvt(Whh0f, Whh0b, 4 * HDIM * HDIM);
  cvt(Wih1f, Wih1b, 4 * HDIM * HDIM);
  cvt(Whh1f, Whh1b, 4 * HDIM * HDIM);
  cvt(x, xbf, T_STEPS * BATCH * INDIM);

  // allow >64KB dynamic LDS (CDNA5: up to 320KB per workgroup)
  (void)hipFuncSetAttribute((const void*)lstm_persistent,
                            hipFuncAttributeMaxDynamicSharedMemorySize,
                            (int)SMEM_BYTES);

  const int nwg = HDIM / 8;  // 64 workgroups, one 8-col hidden slice each
  lstm_persistent<<<nwg, 256, SMEM_BYTES, stream>>>(
      xbf, Wih0b, Whh0b, Wih1b, Whh1b,
      bih0, bhh0, bih1, bhh1,
      h0buf, h1buf, &bar[0], &bar[1],
      (float*)d_out, nwg);
}
